// EncoderDecoder_35905926595426
// MI455X (gfx1250) — matrix operand order repaired and checked by reading, not compiled, not run on hardware
//
#include <hip/hip_runtime.h>

// ---------------------------------------------------------------------------
// Problem dims
// ---------------------------------------------------------------------------
#define S_   49
#define B_   64
#define DENC 512
#define DD   1800
#define DE   512
#define V_   10000
#define T_   20

// padded dims (K padded to mult of 64 for the async K-step, N to mult of 64)
#define KDD_P   1856      // 1800 -> 29*64
#define KCAT_P  2880      // 2824 -> 45*64
#define NDD_P   1856      // 1800 -> 29*64
#define NG_P    7232      // 7200 -> 113*64
#define NV_P    10048     // 10000 -> 157*64

typedef __attribute__((ext_vector_type(16))) __bf16 v16bf;
typedef __attribute__((ext_vector_type(8)))  __bf16 v8bf;
typedef __attribute__((ext_vector_type(8)))  float  v8f;

union FragBF { v16bf v; v8bf h[2]; };

__device__ __forceinline__ float sigm(float x) { return 1.0f / (1.0f + expf(-x)); }

// ---------------------------------------------------------------------------
// Workspace layout (bytes)
// ---------------------------------------------------------------------------
constexpr size_t au(size_t x) { return (x + 255) & ~(size_t)255; }
constexpr size_t OFF_WQ    = 0;                                        // bf16 NDD_P x KDD_P
constexpr size_t OFF_WHA   = au(OFF_WQ    + 2ull * NDD_P * KDD_P);     // bf16 NDD_P x 512
constexpr size_t OFF_WCAT  = au(OFF_WHA   + 2ull * NDD_P * 512);       // bf16 NG_P  x KCAT_P
constexpr size_t OFF_WOUT  = au(OFF_WCAT  + 2ull * NG_P  * KCAT_P);    // bf16 NV_P  x KCAT_P
constexpr size_t OFF_IHW1  = au(OFF_WOUT  + 2ull * NV_P  * KCAT_P);    // bf16 NDD_P x 512
constexpr size_t OFF_IHW2  = au(OFF_IHW1  + 2ull * NDD_P * 512);       // bf16 NDD_P x KDD_P
constexpr size_t OFF_ICW1  = au(OFF_IHW2  + 2ull * NDD_P * KDD_P);     // bf16 NDD_P x 512
constexpr size_t OFF_ICW2  = au(OFF_ICW1  + 2ull * NDD_P * 512);       // bf16 NDD_P x KDD_P
constexpr size_t OFF_HBF   = au(OFF_ICW2  + 2ull * NDD_P * KDD_P);     // bf16 3136 x 512
constexpr size_t OFF_HAVGB = au(OFF_HBF   + 2ull * S_ * B_ * DENC);    // bf16 64 x 512
constexpr size_t OFF_T1B   = au(OFF_HAVGB + 2ull * B_ * DENC);         // bf16 64 x KDD_P
constexpr size_t OFF_HTB   = au(OFF_T1B   + 2ull * B_ * KDD_P);        // bf16 64 x KDD_P
constexpr size_t OFF_XCAT  = au(OFF_HTB   + 2ull * B_ * KDD_P);        // bf16 64 x KCAT_P
constexpr size_t OFF_OCAT  = au(OFF_XCAT  + 2ull * B_ * KCAT_P);       // bf16 64 x KCAT_P
constexpr size_t OFF_HPART = au(OFF_OCAT  + 2ull * B_ * KCAT_P);       // f32 3136 x 1800
constexpr size_t OFF_HAVGF = au(OFF_HPART + 4ull * S_ * B_ * DD);      // f32 64 x 512
constexpr size_t OFF_QPART = au(OFF_HAVGF + 4ull * B_ * DENC);         // f32 64 x 1800
constexpr size_t OFF_U     = au(OFF_QPART + 4ull * B_ * DD);           // f32 49*64
constexpr size_t OFF_CTX   = au(OFF_U     + 4ull * S_ * B_);           // f32 64 x 512
constexpr size_t OFF_GATES = au(OFF_CTX   + 4ull * B_ * DENC);         // f32 64 x 7200
constexpr size_t OFF_HTF   = au(OFF_GATES + 4ull * B_ * 4 * DD);       // f32 64 x 1800
constexpr size_t OFF_CF    = au(OFF_HTF   + 4ull * B_ * DD);           // f32 64 x 1800
constexpr size_t OFF_BIASL = au(OFF_CF    + 4ull * B_ * DD);           // f32 7200
constexpr size_t OFF_TOK   = au(OFF_BIASL + 4ull * 4 * DD);            // int 64
constexpr size_t WS_TOTAL  = au(OFF_TOK   + 4ull * 64);

// ---------------------------------------------------------------------------
// WMMA GEMM: C[M,Nact] = A[M,Kp] (bf16, ld=Kp) @ Bw[Np,Kp]^T (bf16, ld=Kp)
// M mult of 64, Kp mult of 64. Block = 256 threads (8 waves), 64x64 tile.
// Triple-buffered async global->LDS pipeline (ASYNCcnt), one barrier/iter,
// K-step 64 => 4 v_wmma per wave per iteration. Last tile peeled so the
// steady-state loop has an unconditional s_wait_asynccnt 4.
// Epilogue: +bias, optional relu, write f32 (ldC) and/or bf16 (ldCb).
// ---------------------------------------------------------------------------
__global__ __launch_bounds__(256) void k_gemm_bf16(
    const __bf16* __restrict__ A, const __bf16* __restrict__ Bw,
    float* __restrict__ C, int ldC,
    __bf16* __restrict__ Cb, int ldCb,
    const float* __restrict__ bias,
    int Nact, int Kp, int act)
{
  __shared__ __bf16 sA[3][64 * 64];   // 3 x 8KB
  __shared__ __bf16 sB[3][64 * 64];   // 3 x 8KB

  const int tid  = threadIdx.x;
  const int m0   = blockIdx.y * 64;
  const int n0   = blockIdx.x * 64;
  const int wave = tid >> 5;
  const int lane = tid & 31;
  const int mi   = wave & 3;          // M sub-tile (16 rows)
  const int njb  = (wave >> 2) * 2;   // first of two N sub-tiles
  const int r    = lane & 15;
  const int kbA  = (lane >> 4) * 8;   // A frag: 2 halves of 8 at +0/+16
  const int kbB  = (lane >> 4) * 8;   // B frag: 2 halves of 8 at +0/+16

  const int nT = Kp >> 6;             // K tiles of 64

  v8f acc0 = {}; v8f acc1 = {};

  // async-stage one 64x64 A tile + one 64x64 B tile into buffer `buf`
  // (4 async instructions per wave per tile-pair)
  auto issue = [&](int kt, int buf) {
#pragma unroll
    for (int i = 0; i < 2; ++i) {
      int c    = tid + 256 * i;       // 512 chunks of 16B per matrix
      int row  = c >> 3;
      int colg = (c & 7) * 8;
      unsigned lA = (unsigned)(size_t)&sA[buf][row * 64 + colg];
      unsigned lB = (unsigned)(size_t)&sB[buf][row * 64 + colg];
      const __bf16* gA = &A[(size_t)(m0 + row) * Kp + (size_t)kt * 64 + colg];
      const __bf16* gB = &Bw[(size_t)(n0 + row) * Kp + (size_t)kt * 64 + colg];
      asm volatile("global_load_async_to_lds_b128 %0, %1, off"
                   :: "v"(lA), "v"(gA) : "memory");
      asm volatile("global_load_async_to_lds_b128 %0, %1, off"
                   :: "v"(lB), "v"(gB) : "memory");
    }
  };

  auto compute = [&](int buf) {
    const __bf16* Ab = sA[buf];
    const __bf16* Bb = sB[buf];
#pragma unroll
    for (int hh = 0; hh < 2; ++hh) {
      const int ko = hh * 32;
      FragBF fa, fb0, fb1;
      const __bf16* ap = &Ab[(mi * 16 + r) * 64 + ko + kbA];
      fa.h[0] = *(const v8bf*)(ap);
      fa.h[1] = *(const v8bf*)(ap + 16);
      const __bf16* bp0 = &Bb[(njb * 16 + r) * 64 + ko + kbB];
      fb0.h[0] = *(const v8bf*)(bp0);
      fb0.h[1] = *(const v8bf*)(bp0 + 16);
      const __bf16* bp1 = &Bb[((njb + 1) * 16 + r) * 64 + ko + kbB];
      fb1.h[0] = *(const v8bf*)(bp1);
      fb1.h[1] = *(const v8bf*)(bp1 + 16);

      acc0 = __builtin_amdgcn_wmma_f32_16x16x32_bf16(false, fa.v, false, fb0.v,
                                                     (short)0, acc0, false, false);
      acc1 = __builtin_amdgcn_wmma_f32_16x16x32_bf16(false, fa.v, false, fb1.v,
                                                     (short)0, acc1, false, false);
    }
  };

  issue(0, 0);
  if (nT > 1) issue(1, 1);

  // steady state: tiles 0 .. nT-2 (tile kt retires, kt+1 stays in flight)
  int buf = 0;
  for (int kt = 0; kt < nT - 1; ++kt) {
    asm volatile("s_wait_asynccnt 4" ::: "memory");
    __syncthreads();                  // tile kt visible; buf (kt+2)%3 free
    if (kt + 2 < nT) {
      int nb = buf + 2; if (nb >= 3) nb -= 3;
      issue(kt + 2, nb);
    }
    compute(buf);
    ++buf; if (buf == 3) buf = 0;
  }
  // peeled last tile
  asm volatile("s_wait_asynccnt 0" ::: "memory");
  __syncthreads();
  compute(buf);

  const int cc  = lane & 15;
  const int rhi = (lane >> 4) * 8;
  for (int half = 0; half < 2; ++half) {
    v8f acc = half ? acc1 : acc0;
    int col = n0 + (njb + half) * 16 + cc;
    if (col < Nact) {
      float bv = bias ? bias[col] : 0.0f;
#pragma unroll
      for (int v = 0; v < 8; ++v) {
        int row = m0 + mi * 16 + v + rhi;
        float x = acc[v] + bv;
        if (act) x = fmaxf(x, 0.0f);
        if (C)  C[(size_t)row * ldC + col] = x;
        if (Cb) Cb[(size_t)row * ldCb + col] = (__bf16)x;
      }
    }
  }
}

// ---------------------------------------------------------------------------
// Small helper kernels
// ---------------------------------------------------------------------------
__global__ void k_convert(const float* __restrict__ src, int srcLd, int srcOff,
                          __bf16* __restrict__ dst, int dstLd, int N, int K)
{
  int idx = blockIdx.x * 256 + threadIdx.x;
  if (idx >= N * K) return;
  int n = idx / K, k = idx % K;
  dst[(size_t)n * dstLd + k] = (__bf16)src[(size_t)n * srcLd + srcOff + k];
}

__global__ void k_bias_combine(const float* a, const float* b, float* c) {
  int i = blockIdx.x * 256 + threadIdx.x;
  if (i < 4 * DD) c[i] = a[i] + b[i];
}

__global__ void k_init_tok(int* tok) {
  if (threadIdx.x < B_) tok[threadIdx.x] = 0;
}

__global__ void k_havg(const float* __restrict__ h, float* __restrict__ havgf,
                       __bf16* __restrict__ havgb)
{
  int idx = blockIdx.x * 256 + threadIdx.x;
  if (idx >= B_ * DENC) return;
  float s = 0.0f;
  for (int ss = 0; ss < S_; ++ss) s += h[((size_t)ss * B_) * DENC + idx];
  s *= (1.0f / (float)S_);
  havgf[idx] = s;
  havgb[idx] = (__bf16)s;
}

// u[s*64+b] = attn_b2 + sum_j relu(Hpart[s*64+b, j] + qpart[b, j]) * w2[j]
__global__ void k_ured(const float* __restrict__ Hp, const float* __restrict__ qp,
                       const float* __restrict__ w2, const float* __restrict__ b2,
                       float* __restrict__ u)
{
  int gid  = blockIdx.x * 256 + threadIdx.x;
  int wid  = gid >> 5, lane = gid & 31;
  if (wid >= S_ * B_) return;
  int b = wid & (B_ - 1);
  float acc = 0.0f;
  for (int j = lane; j < DD; j += 32) {
    __builtin_prefetch(&Hp[(size_t)wid * DD + j + 512], 0, 3);
    float z = Hp[(size_t)wid * DD + j] + qp[(size_t)b * DD + j];
    if (z > 0.0f) acc += z * w2[j];
  }
  for (int off = 16; off; off >>= 1) acc += __shfl_xor(acc, off, 32);
  if (lane == 0) u[wid] = acc + b2[0];
}

// per-b: beta = sigmoid(ht . beta_w + beta_b); softmax over s; ctx = beta * sum_s a_s h[s,b,:]
__global__ void k_attn_ctx(const float* __restrict__ u, const float* __restrict__ htf,
                           const float* __restrict__ betaW, const float* __restrict__ betaB,
                           const float* __restrict__ h, float* __restrict__ ctx)
{
  __shared__ float red[256];
  __shared__ float a_sh[S_];
  __shared__ float beta_sh;
  int b = blockIdx.x, tid = threadIdx.x;

  float acc = 0.0f;
  for (int j = tid; j < DD; j += 256) acc += htf[(size_t)b * DD + j] * betaW[j];
  red[tid] = acc; __syncthreads();
  for (int off = 128; off; off >>= 1) { if (tid < off) red[tid] += red[tid + off]; __syncthreads(); }
  if (tid == 0) {
    beta_sh = sigm(red[0] + betaB[0]);
    float m = -3.4e38f;
    for (int s = 0; s < S_; ++s) m = fmaxf(m, u[s * B_ + b]);
    float sum = 0.0f;
    for (int s = 0; s < S_; ++s) { float e = expf(u[s * B_ + b] - m); a_sh[s] = e; sum += e; }
    float inv = 1.0f / sum;
    for (int s = 0; s < S_; ++s) a_sh[s] *= inv;
  }
  __syncthreads();
  for (int e = tid; e < DENC; e += 256) {
    float s = 0.0f;
    for (int si = 0; si < S_; ++si) s += a_sh[si] * h[((size_t)si * B_ + b) * DENC + e];
    ctx[(size_t)b * DENC + e] = s * beta_sh;
  }
}

// xcat = [emb[E_t]; ctx; ht; pad]  (bf16, 64 x KCAT_P)
__global__ void k_build_xcat(const float* __restrict__ emb, const int* __restrict__ et,
                             const float* __restrict__ ctx, const __bf16* __restrict__ htb,
                             __bf16* __restrict__ xcat)
{
  int b = blockIdx.x;
  int tok = et[b];
  for (int c = threadIdx.x; c < KCAT_P; c += 256) {
    __bf16 v;
    if (c < DE)                  v = (__bf16)emb[(size_t)tok * DE + c];
    else if (c < DE + DENC)      v = (__bf16)ctx[(size_t)b * DENC + (c - DE)];
    else if (c < DE + DENC + DD) v = htb[(size_t)b * KDD_P + (c - DE - DENC)];
    else                         v = (__bf16)0.0f;
    xcat[(size_t)b * KCAT_P + c] = v;
  }
}

// ocat = [h_new; emb[prev_tok]; ctx; pad]  (bf16, 64 x KCAT_P)
__global__ void k_build_outcat(const float* __restrict__ emb, const int* __restrict__ tok,
                               const float* __restrict__ ctx, const __bf16* __restrict__ htb,
                               __bf16* __restrict__ ocat)
{
  int b = blockIdx.x;
  int tk = tok[b];
  for (int c = threadIdx.x; c < KCAT_P; c += 256) {
    __bf16 v;
    if (c < DD)                   v = htb[(size_t)b * KDD_P + c];
    else if (c < DD + DE)         v = (__bf16)emb[(size_t)tk * DE + (c - DD)];
    else if (c < DD + DE + DENC)  v = (__bf16)ctx[(size_t)b * DENC + (c - DD - DE)];
    else                          v = (__bf16)0.0f;
    ocat[(size_t)b * KCAT_P + c] = v;
  }
}

__global__ void k_lstm(const float* __restrict__ gates, float* __restrict__ c,
                       float* __restrict__ htf, __bf16* __restrict__ htb)
{
  int idx = blockIdx.x * 256 + threadIdx.x;
  if (idx >= B_ * DD) return;
  int b = idx / DD, j = idx % DD;
  const float* g = gates + (size_t)b * 4 * DD;
  float ig = sigm(g[j]);
  float fg = sigm(g[DD + j]);
  float gg = tanhf(g[2 * DD + j]);
  float og = sigm(g[3 * DD + j]);
  float cn = fg * c[idx] + ig * gg;
  float hn = og * tanhf(cn);
  c[idx] = cn;
  htf[idx] = hn;
  htb[(size_t)b * KDD_P + j] = (__bf16)hn;
}

__global__ void k_argmax(const float* __restrict__ logits, int* __restrict__ tok)
{
  __shared__ float sv[256];
  __shared__ int   si[256];
  int b = blockIdx.x, tid = threadIdx.x;
  float best = -3.4e38f; int bi = tid;
  for (int i = tid; i < V_; i += 256) {
    __builtin_prefetch(&logits[(size_t)b * V_ + i + 1024], 0, 3);
    float v = logits[(size_t)b * V_ + i];
    if (v > best) { best = v; bi = i; }
  }
  sv[tid] = best; si[tid] = bi; __syncthreads();
  for (int off = 128; off; off >>= 1) {
    if (tid < off) {
      if (sv[tid + off] > sv[tid] ||
          (sv[tid + off] == sv[tid] && si[tid + off] < si[tid])) {
        sv[tid] = sv[tid + off]; si[tid] = si[tid + off];
      }
    }
    __syncthreads();
  }
  if (tid == 0) tok[b] = si[0];
}

// ---------------------------------------------------------------------------
// Launch
// ---------------------------------------------------------------------------
extern "C" void kernel_launch(void* const* d_in, const int* in_sizes, int n_in,
                              void* d_out, int out_size, void* d_ws, size_t ws_size,
                              hipStream_t stream)
{
  const float* h      = (const float*)d_in[0];
  const int*   E      = (const int*)  d_in[1];
  const float* emb    = (const float*)d_in[2];
  const float* attnW1 = (const float*)d_in[3];
  const float* attnB1 = (const float*)d_in[4];
  const float* attnW2 = (const float*)d_in[5];
  const float* attnB2 = (const float*)d_in[6];
  const float* betaW  = (const float*)d_in[7];
  const float* betaB  = (const float*)d_in[8];
  const float* Wih    = (const float*)d_in[9];
  const float* Whh    = (const float*)d_in[10];
  const float* bih    = (const float*)d_in[11];
  const float* bhh    = (const float*)d_in[12];
  const float* outW   = (const float*)d_in[13];
  const float* outB   = (const float*)d_in[14];
  const float* ihW1   = (const float*)d_in[15];
  const float* ihB1   = (const float*)d_in[16];
  const float* ihW2   = (const float*)d_in[17];
  const float* ihB2   = (const float*)d_in[18];
  const float* icW1   = (const float*)d_in[19];
  const float* icB1   = (const float*)d_in[20];
  const float* icW2   = (const float*)d_in[21];
  const float* icB2   = (const float*)d_in[22];

  float* out = (float*)d_out;
  char*  ws  = (char*)d_ws;

  __bf16* WQ    = (__bf16*)(ws + OFF_WQ);
  __bf16* WHA   = (__bf16*)(ws + OFF_WHA);
  __bf16* WCAT  = (__bf16*)(ws + OFF_WCAT);
  __bf16* WOUT  = (__bf16*)(ws + OFF_WOUT);
  __bf16* IHW1  = (__bf16*)(ws + OFF_IHW1);
  __bf16* IHW2  = (__bf16*)(ws + OFF_IHW2);
  __bf16* ICW1  = (__bf16*)(ws + OFF_ICW1);
  __bf16* ICW2  = (__bf16*)(ws + OFF_ICW2);
  __bf16* HBF   = (__bf16*)(ws + OFF_HBF);
  __bf16* HAVGB = (__bf16*)(ws + OFF_HAVGB);
  __bf16* T1B   = (__bf16*)(ws + OFF_T1B);
  __bf16* HTB   = (__bf16*)(ws + OFF_HTB);
  __bf16* XCAT  = (__bf16*)(ws + OFF_XCAT);
  __bf16* OCAT  = (__bf16*)(ws + OFF_OCAT);
  float*  HPART = (float*)(ws + OFF_HPART);
  float*  HAVGF = (float*)(ws + OFF_HAVGF);
  float*  QPART = (float*)(ws + OFF_QPART);
  float*  U     = (float*)(ws + OFF_U);
  float*  CTX   = (float*)(ws + OFF_CTX);
  float*  GATES = (float*)(ws + OFF_GATES);
  float*  HTF   = (float*)(ws + OFF_HTF);
  float*  CF    = (float*)(ws + OFF_CF);
  float*  BIASL = (float*)(ws + OFF_BIASL);
  int*    TOK   = (int*)(ws + OFF_TOK);

  // zero scratch (padding regions must be 0 for the padded GEMMs)
  hipMemsetAsync(d_ws, 0, WS_TOTAL <= ws_size ? WS_TOTAL : ws_size, stream);

  auto cvt = [&](const float* src, int srcLd, int srcOff, __bf16* dst, int dstLd,
                 int N, int K) {
    int total = N * K;
    k_convert<<<(total + 255) / 256, 256, 0, stream>>>(src, srcLd, srcOff, dst, dstLd, N, K);
  };

  // weights -> bf16 (padded)
  cvt(attnW1, DD + DENC, 0,  WQ,  KDD_P, DD, DD);     // attn_W1[:, :DD]
  cvt(attnW1, DD + DENC, DD, WHA, DENC,  DD, DENC);   // attn_W1[:, DD:]
  cvt(Wih,  DE + DENC, 0, WCAT,        KCAT_P, 4 * DD, DE + DENC);
  cvt(Whh,  DD,        0, WCAT + 1024, KCAT_P, 4 * DD, DD);
  cvt(outW, DD + DE + DENC, 0, WOUT, KCAT_P, V_, DD + DE + DENC);
  cvt(ihW1, DENC, 0, IHW1, DENC,  DD, DENC);
  cvt(ihW2, DD,   0, IHW2, KDD_P, DD, DD);
  cvt(icW1, DENC, 0, ICW1, DENC,  DD, DENC);
  cvt(icW2, DD,   0, ICW2, KDD_P, DD, DD);
  cvt(h,    DENC, 0, HBF,  DENC,  S_ * B_, DENC);

  k_bias_combine<<<(4 * DD + 255) / 256, 256, 0, stream>>>(bih, bhh, BIASL);
  k_init_tok<<<1, 64, 0, stream>>>(TOK);
  k_havg<<<(B_ * DENC + 255) / 256, 256, 0, stream>>>(h, HAVGF, HAVGB);

  auto gemm = [&](const __bf16* A, const __bf16* Bw, float* C, int ldC,
                  __bf16* Cb, int ldCb, const float* bias,
                  int M, int Nact, int Kp, int act) {
    dim3 g((Nact + 63) / 64, M / 64);
    k_gemm_bf16<<<g, 256, 0, stream>>>(A, Bw, C, ldC, Cb, ldCb, bias, Nact, Kp, act);
  };

  // hoisted attention key projection: Hpart[(s,b), j] = h . attn_W1[j, DD:]
  gemm(HBF, WHA, HPART, DD, nullptr, 0, nullptr, S_ * B_, DD, DENC, 0);

  // init MLPs: h0 and c0
  gemm(HAVGB, IHW1, nullptr, 0, T1B, KDD_P, ihB1, B_, DD, DENC, 1);
  gemm(T1B,   IHW2, HTF, DD, HTB, KDD_P, ihB2, B_, DD, KDD_P, 0);
  gemm(HAVGB, ICW1, nullptr, 0, T1B, KDD_P, icB1, B_, DD, DENC, 1);
  gemm(T1B,   ICW2, CF, DD, nullptr, 0, icB2, B_, DD, KDD_P, 0);

  for (int t = 0; t < T_ - 1; ++t) {
    // query part of attention MLP (+ attn_b1)
    gemm(HTB, WQ, QPART, DD, nullptr, 0, attnB1, B_, DD, KDD_P, 0);
    // u[s,b] = relu(Hpart + qpart) . attn_w2 + attn_b2
    k_ured<<<(S_ * B_) / 8, 256, 0, stream>>>(HPART, QPART, attnW2, attnB2, U);
    // softmax over s, context, beta gate
    k_attn_ctx<<<B_, 256, 0, stream>>>(U, HTF, betaW, betaB, h, CTX);
    // x_cat = [emb[E_t]; ctx; ht]
    k_build_xcat<<<B_, 256, 0, stream>>>(emb, E + t * B_, CTX, HTB, XCAT);
    // gates = x_cat @ [W_ih|W_hh]^T + (b_ih + b_hh)
    gemm(XCAT, WCAT, GATES, 4 * DD, nullptr, 0, BIASL, B_, 4 * DD, KCAT_P, 0);
    // LSTM cell update
    k_lstm<<<(B_ * DD + 255) / 256, 256, 0, stream>>>(GATES, CF, HTF, HTB);
    // deep output input = [h_new; emb[prev_tok]; ctx]
    k_build_outcat<<<B_, 256, 0, stream>>>(emb, TOK, CTX, HTB, OCAT);
    // logits -> d_out[t]
    gemm(OCAT, WOUT, out + (size_t)t * B_ * V_, V_, nullptr, 0, outB, B_, V_, KCAT_P, 0);
    // prev_tok = argmax(logits)
    k_argmax<<<B_, 256, 0, stream>>>(out + (size_t)t * B_ * V_, TOK);
  }
}
